// WHTConv2D_84842783965241
// MI455X (gfx1250) — matrix operand order repaired and checked by reading, not compiled, not run on hardware
//
#include <hip/hip_runtime.h>
#include <hip/hip_bf16.h>

// WHT-Conv2D for MI455X (gfx1250), compile-only tuning. Round 3.
//
// Pipeline (B=16, C=64, H=W=224 padded to 256):
//   K1 wht_fwd : per (b,c) slice, f3 = (H256 . pad(x) . H256) * v        -> ws (bf16)
//   K2 chanmix : f3 <- softthresh( W(64x64) @ f3 along channels, T )     (in place)
//   K3 wht_inv : y = crop( H256 . f3 . H256 ) / 65536 + x                -> d_out
//
// All matmuls use V_WMMA_F32_16X16X32_BF16. Hadamard operands (exact +-1 in
// bf16) are built in registers once per tile-row/column pair (one popcount
// per packed dword). Each wave keeps TWO accumulators live so every
// ds_load_b128 operand feeds two WMMAs (DS:WMMA = 1:1).
// Intermediate kept as bf16 (128 MB in d_ws) -> total HBM traffic ~1.1 GB,
// ~50us at 23.3 TB/s; math (~146 GFLOP) is negligible on the matrix units.
// Each WHT workgroup uses 264 KB dynamic LDS (needs CDNA5's 320KB/WGP).

typedef __attribute__((ext_vector_type(16))) __bf16 v16bf;
typedef __attribute__((ext_vector_type(8)))  __bf16 v8bf;
typedef __attribute__((ext_vector_type(8)))  float  v8f;
typedef __attribute__((ext_vector_type(8)))  unsigned int u32x8;

struct B32B { uint4 lo, hi; };   // 32 bytes -> one v16bf operand
static_assert(sizeof(B32B) == 32, "");
static_assert(sizeof(v16bf) == 32, "");
static_assert(sizeof(u32x8) == 32, "");

#define LDS_STRIDE 264   // 256+8 halves: rows stay 16B aligned, banks staggered

// Build A-operand (16x32 bf16) of the Hadamard block H[rbase+M][kt*32+koff].
// A layout used here (contiguous k): lane L -> M = L%16; lane-half lh = L/16;
// element e -> koff = lh*16 + e. Dword d holds elements 2d,2d+1
// whose K are consecutive with K even -> one popcount per dword:
//   parity(row & (k+1)) = parity(row & k) ^ (row & 1).
__device__ __forceinline__ v16bf had_A(int rbase, int kt, int lm, int lh) {
  u32x8 u;
  const int row = rbase + lm;
  const unsigned rl = (unsigned)row & 1u;
#pragma unroll
  for (int d = 0; d < 8; ++d) {
    const int e = 2 * d;
    const int k0 = kt * 32 + lh * 16 + e;   // even
    const unsigned p0 = (unsigned)__popc(row & k0) & 1u;
    const unsigned p1 = p0 ^ rl;
    u[d] = 0x3F803F80u | (p0 << 15) | (p1 << 31);            // {+-1, +-1} bf16
  }
  return __builtin_bit_cast(v16bf, u);
}

// Build B-operand (32x16 bf16) of H[kt*32+K][cbase+n].
// B layout used here (contiguous k, as in had_A): lane L -> n = L%16;
// element e -> K = (L/16)*16 + e. Same even-K pairing trick.
__device__ __forceinline__ v16bf had_B(int cbase, int kt, int lm, int lh) {
  u32x8 u;
  const int col = cbase + lm;
  const unsigned cl = (unsigned)col & 1u;
#pragma unroll
  for (int d = 0; d < 8; ++d) {
    const int k0 = kt * 32 + lh * 16 + 2 * d;                // even
    const unsigned p0 = (unsigned)__popc(k0 & col) & 1u;
    const unsigned p1 = p0 ^ cl;
    u[d] = 0x3F803F80u | (p0 << 15) | (p1 << 31);
  }
  return __builtin_bit_cast(v16bf, u);
}

// Load A-operand (16x32) from a row-major bf16 LDS matrix with LDS_STRIDE.
// Row = rbase + (L%16); elements 0..15 -> K = kt*32+lh*16 .. +15.
__device__ __forceinline__ v16bf lds_A(const __bf16* m, int rbase, int kt,
                                       int lm, int lh) {
  const char* p = (const char*)&m[(rbase + lm) * LDS_STRIDE + kt * 32 + lh * 16];
  B32B t;
  t.lo = *(const uint4*)p;          // ds_load_b128
  t.hi = *(const uint4*)(p + 16);   // ds_load_b128
  return __builtin_bit_cast(v16bf, t);
}

// Load B-operand (32x16) from LDS holding the matrix TRANSPOSED
// (mT[n][k] = m[k][n], row-major, LDS_STRIDE): lane n = L%16 reads
// 16 consecutive K at kt*32 + (L/16)*16.
__device__ __forceinline__ v16bf lds_Bt(const __bf16* mT, int cbase, int kt,
                                        int lm, int lh) {
  const char* p = (const char*)&mT[(cbase + lm) * LDS_STRIDE + kt * 32 + lh * 16];
  B32B t;
  t.lo = *(const uint4*)p;          // ds_load_b128
  t.hi = *(const uint4*)(p + 16);   // ds_load_b128
  return __builtin_bit_cast(v16bf, t);
}

#define WMMA_BF16(a, b, c) \
  __builtin_amdgcn_wmma_f32_16x16x32_bf16(false, (a), false, (b), (short)0, (c), false, false)

// ---------------------------------------------------------------------------
// K1: forward 2D WHT + v scaling.  grid = B*C (1024), block = 256 (8 waves).
// Dynamic LDS: Xt (X transposed) + T1 (H.X), each 256 x LDS_STRIDE bf16.
// ---------------------------------------------------------------------------
__global__ __launch_bounds__(256) void wht_fwd(const float* __restrict__ x,
                                               const float* __restrict__ v,
                                               __bf16* __restrict__ f3) {
  extern __shared__ char smem[];
  __bf16* Xt = (__bf16*)smem;                              // Xt[w][h] = X[h][w]
  __bf16* T1 = (__bf16*)(smem + 256 * LDS_STRIDE * 2);     // T1[i][k] = (H.X)[i][k]
  const int slice = blockIdx.x;                            // b*64 + c
  const int tid = threadIdx.x;

  // Zero Xt (covers the 224->256 zero padding).
  {
    uint4 z = {0u, 0u, 0u, 0u};
    uint4* zp = (uint4*)Xt;
    for (int i = tid; i < (256 * LDS_STRIDE * 2) / 16; i += 256) zp[i] = z;
  }
  __syncthreads();

  // Stage x slice transposed into LDS as bf16 (float4 loads; 224 = 4*56).
  {
    const float* xs = x + (size_t)slice * (224 * 224);
    for (int i4 = tid; i4 < (224 * 224) / 4; i4 += 256) {
      int idx = i4 * 4;
      int h = idx / 224, w = idx - h * 224;
      float4 d = *(const float4*)(xs + idx);               // global_load_b128
      Xt[(w + 0) * LDS_STRIDE + h] = (__bf16)d.x;
      Xt[(w + 1) * LDS_STRIDE + h] = (__bf16)d.y;
      Xt[(w + 2) * LDS_STRIDE + h] = (__bf16)d.z;
      Xt[(w + 3) * LDS_STRIDE + h] = (__bf16)d.w;
    }
  }
  __syncthreads();

  const int wave = tid >> 5, lane = tid & 31;
  const int lh = lane >> 4, lm = lane & 15;

  // Step 1: T1 = H . X. Each wave owns tile-rows ti0=2*wave, ti1=2*wave+1.
  // Hadamard A-operands hoisted; each B load feeds both accumulators.
  {
    const int ti0 = wave * 2, ti1 = wave * 2 + 1;
    v16bf A0[8], A1[8];
#pragma unroll
    for (int kt = 0; kt < 8; ++kt) {
      A0[kt] = had_A(ti0 * 16, kt, lm, lh);
      A1[kt] = had_A(ti1 * 16, kt, lm, lh);
    }
    for (int tj = 0; tj < 16; ++tj) {
      v8f acc0 = {}, acc1 = {};
#pragma unroll
      for (int kt = 0; kt < 8; ++kt) {
        v16bf b = lds_Bt(Xt, tj * 16, kt, lm, lh);
        acc0 = WMMA_BF16(A0[kt], b, acc0);
        acc1 = WMMA_BF16(A1[kt], b, acc1);
      }
      // C/D layout: lane -> (M = r + 8*lh, N = lm)
#pragma unroll
      for (int r = 0; r < 8; ++r) {
        T1[(ti0 * 16 + r + 8 * lh) * LDS_STRIDE + tj * 16 + lm] = (__bf16)acc0[r];
        T1[(ti1 * 16 + r + 8 * lh) * LDS_STRIDE + tj * 16 + lm] = (__bf16)acc1[r];
      }
    }
  }
  __syncthreads();

  // Step 2: f3 = (T1 . H) * v. Each wave owns tile-cols tj0=2*wave, tj1=+1;
  // Hadamard B-operands hoisted; each A load feeds both accumulators.
  __bf16* out = f3 + (size_t)slice * 65536;
  {
    const int tj0 = wave * 2, tj1 = wave * 2 + 1;
    v16bf B0[8], B1[8];
#pragma unroll
    for (int kt = 0; kt < 8; ++kt) {
      B0[kt] = had_B(tj0 * 16, kt, lm, lh);
      B1[kt] = had_B(tj1 * 16, kt, lm, lh);
    }
    const int j0 = tj0 * 16 + lm, j1 = tj1 * 16 + lm;
    for (int ti = 0; ti < 16; ++ti) {
      v8f acc0 = {}, acc1 = {};
#pragma unroll
      for (int kt = 0; kt < 8; ++kt) {
        v16bf a = lds_A(T1, ti * 16, kt, lm, lh);
        acc0 = WMMA_BF16(a, B0[kt], acc0);
        acc1 = WMMA_BF16(a, B1[kt], acc1);
      }
#pragma unroll
      for (int r = 0; r < 8; ++r) {
        int i = ti * 16 + r + 8 * lh;
        out[i * 256 + j0] = (__bf16)(acc0[r] * v[i * 256 + j0]);
        out[i * 256 + j1] = (__bf16)(acc1[r] * v[i * 256 + j1]);
      }
    }
  }
}

// ---------------------------------------------------------------------------
// K2: channel mix (64x64 GEMM over channels) + soft-threshold, in place.
// grid = B * 256 spatial chunks (4096), block = 256. Each WG owns
// f3[b, 0..63, sbase..sbase+255]; stages it transposed in LDS, so the
// in-place global writeback is race-free.
// ---------------------------------------------------------------------------
__global__ __launch_bounds__(256) void chanmix(const float* __restrict__ Wm,
                                               const float* __restrict__ T,
                                               __bf16* __restrict__ f3) {
  __shared__ __bf16 Lt[256 * 64];   // Lt[s_local][c]  (32 KB)
  __shared__ __bf16 Wl[64 * 64];    // Wl[o][c]        (8 KB)
  const int b = blockIdx.x >> 8;
  const int sbase = (blockIdx.x & 255) * 256;
  const int tid = threadIdx.x;

  for (int i = tid; i < 64 * 64; i += 256) Wl[i] = (__bf16)Wm[i];
  {
    int c = tid >> 2, part = tid & 3;                       // 4 threads per channel row
    const __bf16* src = f3 + (size_t)(b * 64 + c) * 65536 + sbase + part * 64;
#pragma unroll
    for (int q = 0; q < 8; ++q) {
      uint4 d = *(const uint4*)(src + q * 8);               // global_load_b128
      v8bf dv = __builtin_bit_cast(v8bf, d);
      int s0 = part * 64 + q * 8;
#pragma unroll
      for (int e = 0; e < 8; ++e) Lt[(s0 + e) * 64 + c] = dv[e];
    }
  }
  __syncthreads();

  const int wave = tid >> 5, lane = tid & 31;
  const int lh = lane >> 4, lm = lane & 15;

  // 4 o-tiles x 16 s-subtiles; W-operands hoisted per o-tile (K=64 -> 2 WMMA).
  for (int ot = 0; ot < 4; ++ot) {
    v16bf A0, A1;
    {
      const char* ap = (const char*)&Wl[(ot * 16 + lm) * 64 + lh * 16];
      B32B t0; t0.lo = *(const uint4*)ap;        t0.hi = *(const uint4*)(ap + 16);
      B32B t1; t1.lo = *(const uint4*)(ap + 64); t1.hi = *(const uint4*)(ap + 80);
      A0 = __builtin_bit_cast(v16bf, t0);
      A1 = __builtin_bit_cast(v16bf, t1);
    }
    for (int sub = wave; sub < 16; sub += 8) {
      v8f acc = {};
      const char* bp = (const char*)&Lt[(sub * 16 + lm) * 64 + lh * 16];
      B32B bt0; bt0.lo = *(const uint4*)bp;        bt0.hi = *(const uint4*)(bp + 16);
      B32B bt1; bt1.lo = *(const uint4*)(bp + 64); bt1.hi = *(const uint4*)(bp + 80);
      acc = WMMA_BF16(A0, __builtin_bit_cast(v16bf, bt0), acc);
      acc = WMMA_BF16(A1, __builtin_bit_cast(v16bf, bt1), acc);

      int j = sbase + sub * 16 + lm;              // global spatial index (same for all r)
      float thr = fmaxf(T[j], 0.0f);              // relu(T)
#pragma unroll
      for (int r = 0; r < 8; ++r) {
        int o = ot * 16 + r + 8 * lh;
        float f = acc[r];
        float av = fabsf(f) - thr;
        float res = av > 0.0f ? copysignf(av, f) : 0.0f;
        f3[(size_t)(b * 64 + o) * 65536 + j] = (__bf16)res; // in place
      }
    }
  }
}

// ---------------------------------------------------------------------------
// K3: inverse 2D WHT, scale 1/65536, crop to 224x224, add residual x.
// grid = B*C (1024), block = 256, dynamic LDS like K1. Cropping: only
// tile-rows/cols 0..13 (i,j < 224) are ever needed -> waves 0..6 compute
// pairs, wave 7 idles in the matmul loops (barriers stay uniform).
// ---------------------------------------------------------------------------
__global__ __launch_bounds__(256) void wht_inv(const __bf16* __restrict__ f6,
                                               const float* __restrict__ x,
                                               float* __restrict__ y) {
  extern __shared__ char smem[];
  __bf16* Xt = (__bf16*)smem;
  __bf16* T1 = (__bf16*)(smem + 256 * LDS_STRIDE * 2);
  const int slice = blockIdx.x;
  const int tid = threadIdx.x;

  // Stage f6 slice transposed (uint4 = 8 bf16 per load; rows are 256 wide).
  {
    const __bf16* fs = f6 + (size_t)slice * 65536;
    for (int i8 = tid; i8 < 65536 / 8; i8 += 256) {
      int idx = i8 * 8;
      int h = idx >> 8, w = idx & 255;
      uint4 d = *(const uint4*)(fs + idx);                  // global_load_b128
      v8bf dv = __builtin_bit_cast(v8bf, d);
#pragma unroll
      for (int e = 0; e < 8; ++e) Xt[(w + e) * LDS_STRIDE + h] = dv[e];
    }
  }
  __syncthreads();

  const int wave = tid >> 5, lane = tid & 31;
  const int lh = lane >> 4, lm = lane & 15;

  // Step 1: T1 = H . f6, rows 0..223 only (ti pairs 0..13 on waves 0..6).
  if (wave < 7) {
    const int ti0 = wave * 2, ti1 = wave * 2 + 1;
    v16bf A0[8], A1[8];
#pragma unroll
    for (int kt = 0; kt < 8; ++kt) {
      A0[kt] = had_A(ti0 * 16, kt, lm, lh);
      A1[kt] = had_A(ti1 * 16, kt, lm, lh);
    }
    for (int tj = 0; tj < 16; ++tj) {
      v8f acc0 = {}, acc1 = {};
#pragma unroll
      for (int kt = 0; kt < 8; ++kt) {
        v16bf b = lds_Bt(Xt, tj * 16, kt, lm, lh);
        acc0 = WMMA_BF16(A0[kt], b, acc0);
        acc1 = WMMA_BF16(A1[kt], b, acc1);
      }
#pragma unroll
      for (int r = 0; r < 8; ++r) {
        T1[(ti0 * 16 + r + 8 * lh) * LDS_STRIDE + tj * 16 + lm] = (__bf16)acc0[r];
        T1[(ti1 * 16 + r + 8 * lh) * LDS_STRIDE + tj * 16 + lm] = (__bf16)acc1[r];
      }
    }
  }
  __syncthreads();

  // Step 2: y = crop(T1 . H) / 65536 + x. tj,ti < 14 -> i,j < 224 always,
  // so the stores need no per-element guards.
  const float* xs = x + (size_t)slice * (224 * 224);
  float* ys = y + (size_t)slice * (224 * 224);
  const float inv = 1.0f / 65536.0f;
  if (wave < 7) {
    const int tj0 = wave * 2, tj1 = wave * 2 + 1;
    v16bf B0[8], B1[8];
#pragma unroll
    for (int kt = 0; kt < 8; ++kt) {
      B0[kt] = had_B(tj0 * 16, kt, lm, lh);
      B1[kt] = had_B(tj1 * 16, kt, lm, lh);
    }
    const int j0 = tj0 * 16 + lm, j1 = tj1 * 16 + lm;
    for (int ti = 0; ti < 14; ++ti) {
      v8f acc0 = {}, acc1 = {};
#pragma unroll
      for (int kt = 0; kt < 8; ++kt) {
        v16bf a = lds_A(T1, ti * 16, kt, lm, lh);
        acc0 = WMMA_BF16(a, B0[kt], acc0);
        acc1 = WMMA_BF16(a, B1[kt], acc1);
      }
#pragma unroll
      for (int r = 0; r < 8; ++r) {
        int i = ti * 16 + r + 8 * lh;
        int idx = i * 224 + j0;
        ys[idx]              = acc0[r] * inv + xs[idx];
        ys[idx + (j1 - j0)]  = acc1[r] * inv + xs[idx + (j1 - j0)];
      }
    }
  }
}

extern "C" void kernel_launch(void* const* d_in, const int* in_sizes, int n_in,
                              void* d_out, int out_size, void* d_ws, size_t ws_size,
                              hipStream_t stream) {
  const float* x  = (const float*)d_in[0];   // (16,64,224,224)
  const float* v  = (const float*)d_in[1];   // (256,256)
  const float* T  = (const float*)d_in[2];   // (256,256)
  const float* Wm = (const float*)d_in[3];   // (64,64)
  float* y = (float*)d_out;
  __bf16* f3 = (__bf16*)d_ws;                // 16*64*65536 bf16 = 128 MB

  const size_t ldsWHT = (size_t)2 * 256 * LDS_STRIDE * sizeof(__bf16); // 264 KB
  wht_fwd<<<16 * 64, 256, ldsWHT, stream>>>(x, v, f3);
  chanmix<<<16 * 256, 256, 0, stream>>>(Wm, T, f3);
  wht_inv<<<16 * 64, 256, ldsWHT, stream>>>(f3, x, y);
}
